// LSTMpredictor_44976897523784
// MI455X (gfx1250) — matrix operand order repaired and checked by reading, not compiled, not run on hardware
//
#include <hip/hip_runtime.h>
#include <hip/hip_bf16.h>

typedef __attribute__((ext_vector_type(16))) __bf16 v16bf;
typedef __attribute__((ext_vector_type(8)))  __bf16 v8bf;
typedef __attribute__((ext_vector_type(8)))  float  v8f;

// Problem constants (match reference)
constexpr int kB    = 2048;
constexpr int kT    = 2048;
constexpr int kH    = 51;
constexpr int kG    = 204;          // 4*H
constexpr int kFUT  = 128;
constexpr int kS    = kT + kFUT;    // 2176 total outputs
constexpr int kGP   = 256;          // gates padded to 16 N-tiles
constexpr int kKP   = 64;           // K padded (two K=32 WMMA chunks)
constexpr int kGS   = 260;          // gates LDS row stride (floats), bank-friendly
constexpr int kHS   = 72;           // hbuf row stride (halfs), bank-friendly
constexpr int kFCR  = 204;          // fc_w folded in as gate row 204

// ---------------------------------------------------------------------------
// Pack W_hh (204x51) + fc_w (51) into padded bf16 weight matrix Wp[256][64].
// ---------------------------------------------------------------------------
__global__ void setup_weights(const float* __restrict__ W_hh,
                              const float* __restrict__ fc_w,
                              __bf16* __restrict__ Wp) {
  int g = threadIdx.x;  // 0..255
  for (int k = 0; k < kKP; ++k) {
    float v = 0.f;
    if (g < kG)          { if (k < kH) v = W_hh[g * kH + k]; }
    else if (g == kFCR)  { if (k < kH) v = fc_w[k]; }
    Wp[g * kKP + k] = (__bf16)v;
  }
}

// ---------------------------------------------------------------------------
// Persistent LSTM: 1 WG = 16 batch rows, 8 waves, weights live in registers.
// ---------------------------------------------------------------------------
__global__ __launch_bounds__(256) void lstm_persistent(
    const float*  __restrict__ x,     // [B, T]
    const float*  __restrict__ W_ih,  // [204]
    const float*  __restrict__ b_ih,  // [204]
    const float*  __restrict__ b_hh,  // [204]
    const float*  __restrict__ fc_b,  // [1]
    const __bf16* __restrict__ Wp,    // [256, 64] packed bf16
    float*        __restrict__ out)   // [B, 2176]
{
  __shared__ __attribute__((aligned(16))) __bf16 hbuf[16 * kHS]; // h_{s-1}, bf16, A-layout friendly
  __shared__ float gates[16 * kGS];   // raw WMMA results (h @ Wp^T)
  __shared__ float xbuf[16 * 64];     // staged x block
  __shared__ float obuf[16 * 64];     // staged out block
  __shared__ float bsh[kGP];          // b_ih + b_hh (padded)
  __shared__ float wih[kGP];          // W_ih column (padded)

  const int tid  = threadIdx.x;
  const int lane = tid & 31;
  const int wave = tid >> 5;            // 0..7
  const int b0   = blockIdx.x * 16;     // batch tile base

  // --- init LDS constants / zero h ---
  {
    float bb = 0.f, wv = 0.f;
    if (tid < kG) { bb = b_ih[tid] + b_hh[tid]; wv = W_ih[tid]; }
    bsh[tid] = bb; wih[tid] = wv;
  }
  for (int i = tid; i < 16 * kHS; i += 256) hbuf[i] = (__bf16)0.f;
  const float fcb = fc_b[0];

  // --- load persistent B-fragments: wave owns N-tiles 2w, 2w+1 ---
  // B layout (32x16 bf16): lanes 0-15 -> N=lane, K{0..7,16..23}; lanes 16-31 -> N=lane-16, K{8..15,24..31}
  v16bf wfrag[2][2];  // [n-tile][k-chunk]
  #pragma unroll
  for (int tn = 0; tn < 2; ++tn) {
    const int col = (wave * 2 + tn) * 16 + (lane & 15);
    #pragma unroll
    for (int c = 0; c < 2; ++c) {
      const int k0 = c * 32 + ((lane >> 4) << 3);
      const v8bf w0 = *(const v8bf*)(Wp + col * kKP + k0);
      const v8bf w1 = *(const v8bf*)(Wp + col * kKP + k0 + 16);
      wfrag[tn][c] = __builtin_shufflevector(w0, w1, 0, 1, 2, 3, 4, 5, 6, 7, 8, 9, 10, 11, 12, 13, 14, 15);
    }
  }

  // --- elementwise (m,j) assignment: 16*51=816 elems over 256 threads ---
  int  em[4], ej[4];
  bool ev[4];
  #pragma unroll
  for (int q = 0; q < 4; ++q) {
    const int e = tid + q * 256;
    ev[q] = (e < 16 * kH);
    em[q] = e / kH;
    ej[q] = e % kH;
  }
  float creg[4] = {0.f, 0.f, 0.f, 0.f};  // fp32 cell state in registers

  __syncthreads();

  for (int s = 0; s <= kS; ++s) {
    // stage next 64 inputs (coalesced) at block boundaries
    if ((s & 63) == 0 && s < kT) {
      const int m = tid >> 4, tl = tid & 15;
      #pragma unroll
      for (int q = 0; q < 4; ++q)
        xbuf[m * 64 + tl + q * 16] = x[(size_t)(b0 + m) * kT + s + tl + q * 16];
    }

    // A fragments of h_{s-1} from LDS
    // A layout (16x32 bf16): lanes 0-15 -> M=lane, halfs = K{0..7,16..23}; lanes 16-31 -> K{8..15,24..31}
    v16bf afrag[2];
    {
      const int m = lane & 15, lo = (lane >> 4) << 3;
      #pragma unroll
      for (int c = 0; c < 2; ++c) {
        const v8bf a0 = *(const v8bf*)(hbuf + m * kHS + c * 32 + lo);
        const v8bf a1 = *(const v8bf*)(hbuf + m * kHS + c * 32 + 16 + lo);
        #pragma unroll
        for (int i = 0; i < 8; ++i) { afrag[c][i] = a0[i]; afrag[c][8 + i] = a1[i]; }
      }
    }

    // gates_raw = h_{s-1} @ Wp^T  (fp32 accumulate)
    #pragma unroll
    for (int tn = 0; tn < 2; ++tn) {
      v8f acc = {0.f, 0.f, 0.f, 0.f, 0.f, 0.f, 0.f, 0.f};
      acc = __builtin_amdgcn_wmma_f32_16x16x32_bf16(false, afrag[0], false, wfrag[tn][0],
                                                    (short)0, acc, false, false);
      acc = __builtin_amdgcn_wmma_f32_16x16x32_bf16(false, afrag[1], false, wfrag[tn][1],
                                                    (short)0, acc, false, false);
      // C layout: lanes 0-15 -> N=lane, M=r ; lanes 16-31 -> N=lane-16, M=8+r
      const int n  = (wave * 2 + tn) * 16 + (lane & 15);
      const int mb = (lane >> 4) << 3;
      #pragma unroll
      for (int r = 0; r < 8; ++r) gates[(mb + r) * kGS + n] = acc[r];
    }
    __syncthreads();

    // out[t=s-1] = h_{s-1} @ fc_w + fc_b  (folded gate row 204)
    if (s >= 1 && tid < 16)
      obuf[tid * 64 + ((s - 1) & 63)] = gates[tid * kGS + kFCR] + fcb;

    // state update to (h_s, c_s); autoregressive input for s >= T
    if (s < kS) {
      #pragma unroll
      for (int q = 0; q < 4; ++q) {
        if (ev[q]) {
          const int m = em[q], j = ej[q];
          const float inp = (s < kT) ? xbuf[m * 64 + (s & 63)]
                                     : (gates[m * kGS + kFCR] + fcb);
          const float gi = gates[m * kGS +       j] + inp * wih[      j] + bsh[      j];
          const float gf = gates[m * kGS +  51 + j] + inp * wih[ 51 + j] + bsh[ 51 + j];
          const float gg = gates[m * kGS + 102 + j] + inp * wih[102 + j] + bsh[102 + j];
          const float go = gates[m * kGS + 153 + j] + inp * wih[153 + j] + bsh[153 + j];
          const float si = 1.f / (1.f + __expf(-gi));
          const float sf = 1.f / (1.f + __expf(-gf));
          const float so = 1.f / (1.f + __expf(-go));
          const float tg = 1.f - 2.f / (1.f + __expf(2.f * gg));
          const float cn = sf * creg[q] + si * tg;
          creg[q] = cn;
          const float tc = 1.f - 2.f / (1.f + __expf(2.f * cn));
          hbuf[m * kHS + j] = (__bf16)(so * tc);
        }
      }
    }
    __syncthreads();

    // flush 64 finished outputs (coalesced); kS % 64 == 0 so this covers everything
    if (s > 0 && (s & 63) == 0) {
      const int m = tid >> 4, tl = tid & 15, tb = s - 64;
      #pragma unroll
      for (int q = 0; q < 4; ++q)
        out[(size_t)(b0 + m) * kS + tb + tl + q * 16] = obuf[m * 64 + tl + q * 16];
    }
  }
}

// ---------------------------------------------------------------------------
extern "C" void kernel_launch(void* const* d_in, const int* in_sizes, int n_in,
                              void* d_out, int out_size, void* d_ws, size_t ws_size,
                              hipStream_t stream) {
  const float* x    = (const float*)d_in[0];
  const float* W_ih = (const float*)d_in[1];   // [204,1]
  const float* W_hh = (const float*)d_in[2];   // [204,51]
  const float* b_ih = (const float*)d_in[3];
  const float* b_hh = (const float*)d_in[4];
  const float* fc_w = (const float*)d_in[5];   // [1,51]
  const float* fc_b = (const float*)d_in[6];
  (void)in_sizes; (void)n_in; (void)out_size; (void)ws_size;

  __bf16* Wp = (__bf16*)d_ws;  // 256*64*2 = 32 KB scratch

  setup_weights<<<1, 256, 0, stream>>>(W_hh, fc_w, Wp);
  lstm_persistent<<<kB / 16, 256, 0, stream>>>(x, W_ih, b_ih, b_hh, fc_b, Wp,
                                               (float*)d_out);
}
